// TileInit_60112362274870
// MI455X (gfx1250) — compile-verified, matrix operand order checked by reading, not run on hardware
//
#include <hip/hip_runtime.h>

// ---------- types ----------
typedef __attribute__((ext_vector_type(16))) _Float16 v16h;
typedef __attribute__((ext_vector_type(8)))  _Float16 v8h;
typedef __attribute__((ext_vector_type(8)))  float    v8f;

union Frag16 { v16h v; v8h h[2]; _Float16 e[16]; };
union Frag8  { v8f  v; float e[8]; };

#define Bn 2
#define Hh 160
#define Wn 320
#define Dd 24
#define HWn (Hh*Wn)
// padded dims (1-cell halo in d,h,w)
#define Dp 26
#define Hp 162
#define Wp 322
#define SP3 (Dp*Hp*Wp)   // padded 3D spatial size
#define SP2 (Hp*Wp)      // padded 2D spatial size

// stats layout (floats): [stage(4)][bg(16)][sum,ssq] at 0..127 ; [stage][bg][mean,rstd] at 128..255

__device__ __forceinline__ int kmap(int i, int lane){
  return (i < 8 ? i : i + 8) + ((lane & 16) ? 8 : 0);
}
__device__ __forceinline__ float silu_f(float x){ return x / (1.f + __expf(-x)); }
__device__ __forceinline__ int toff3(int t){ return ((t/9)*Hp + ((t%9)/3))*Wp + (t%3); }
__device__ __forceinline__ int toff2(int t){ return (t/3)*Wp + (t%3); }

// ---------- misc small kernels ----------
__global__ void k_zero(float* p, int n){
  int i = blockIdx.x * 256 + threadIdx.x;
  if (i < n) p[i] = 0.f;
}
__global__ void k_fill32(unsigned int* p, int n){
  int i = blockIdx.x * 256 + threadIdx.x;
  if (i < n) p[i] = 0u;
}

__global__ void k_finalize(float* stats){
  int t = threadIdx.x;
  if (t >= 64) return;
  int stage = t >> 4, bg = t & 15;
  float cnt = (stage < 2) ? 2457600.f : (stage == 2 ? 102400.f : 204800.f);
  float s = stats[stage*32 + bg*2 + 0];
  float q = stats[stage*32 + bg*2 + 1];
  float mean = s / cnt;
  float var  = q / cnt - mean * mean;
  if (var < 0.f) var = 0.f;
  stats[128 + stage*32 + bg*2 + 0] = mean;
  stats[128 + stage*32 + bg*2 + 1] = rsqrtf(var + 1e-5f);
}

// ---------- weight repack into WMMA A-fragment order ----------
// conv1: chunk c = kernel row (kd*3+kh), tsub = kw (0..2; 3 = zero pad)
__global__ void k_pack_w1(const float* __restrict__ w, _Float16* __restrict__ ap){
  for (int idx = threadIdx.x; idx < 9*32*16; idx += blockDim.x){
    int i = idx & 15, lane = (idx >> 4) & 31, c = idx >> 9;
    int k = kmap(i, lane); int m = lane & 15;
    int ci = k & 7, kw = k >> 3;
    ap[idx] = (kw < 3) ? (_Float16)w[(m*8 + ci)*27 + c*3 + kw] : (_Float16)0.f;
  }
}
__global__ void k_pack_w2(const float* __restrict__ w, _Float16* __restrict__ ap){
  for (int idx = threadIdx.x; idx < 14*32*16; idx += blockDim.x){
    int i = idx & 15, lane = (idx >> 4) & 31, c = idx >> 9;
    int k = kmap(i, lane); int m = lane & 15;
    int ci = k & 15, t = c*2 + (k >> 4);
    ap[idx] = (t < 27) ? (_Float16)w[(m*16 + ci)*27 + t] : (_Float16)0.f;
  }
}
__global__ void k_pack_wf(const float* __restrict__ w, _Float16* __restrict__ ap){
  for (int idx = threadIdx.x; idx < 18*32*16; idx += blockDim.x){
    int i = idx & 15, lane = (idx >> 4) & 31, c = idx >> 9;
    int k = kmap(i, lane); int m = lane & 15;
    int t = c >> 1, ci = (c & 1)*32 + k;
    ap[idx] = (_Float16)w[(m*64 + ci)*9 + t];
  }
}

// ---------- transpose fL -> padded channel-last f16 [B][Hp][Wp][64] (interior only) ----------
__global__ void k_padLt(const float* __restrict__ fL, _Float16* __restrict__ dst){
  __shared__ _Float16 tile[64][65];
  int b = blockIdx.z, h = blockIdx.y, w0 = blockIdx.x * 64;
  int tid = threadIdx.x;
  for (int idx = tid; idx < 64*64; idx += 256){
    int c = idx >> 6, j = idx & 63;
    tile[c][j] = (_Float16)fL[((b*64 + c)*Hh + h)*Wn + w0 + j];
  }
  __syncthreads();
  for (int idx = tid; idx < 64*64; idx += 256){
    int j = idx >> 6, c = idx & 63;                    // consecutive tid -> consecutive c
    dst[((b*Hp + h+1)*Wp + (w0 + j + 1))*64 + c] = tile[c][j];
  }
}

// ---------- cost volume: all 8 groups per block, v8h stores, padded channel-last ----------
__global__ void k_cost(const float* __restrict__ fL, const float* __restrict__ fR,
                       _Float16* __restrict__ cvp){
  __shared__ float sL[64][65];
  __shared__ float sR[64][89];
  int b = blockIdx.z, h = blockIdx.y, w0 = blockIdx.x * 64;
  int tid = threadIdx.x;
  for (int idx = tid; idx < 64*64; idx += 256){
    int c = idx >> 6, j = idx & 63;
    sL[c][j] = fL[((b*64 + c)*Hh + h)*Wn + w0 + j];
  }
  for (int idx = tid; idx < 64*88; idx += 256){
    int c = idx / 88, j = idx % 88;
    int ww = w0 - 24 + j;
    sR[c][j] = (ww >= 0) ? fR[((b*64 + c)*Hh + h)*Wn + ww] : 0.f;
  }
  __syncthreads();
  for (int o = tid; o < 24*64; o += 256){
    int d = o >> 6, j = o & 63;
    int jr = j + 24 - d;
    v8h pk;
    #pragma unroll
    for (int g = 0; g < 8; ++g){
      float a = 0.f;
      #pragma unroll
      for (int ci = 0; ci < 8; ++ci) a += sL[g*8+ci][j] * sR[g*8+ci][jr];
      pk[g] = (_Float16)(a * 0.125f);
    }
    *(v8h*)(cvp + (((b*Dp + d+1)*Hp + h+1)*Wp + (w0 + j + 1))*8) = pk;
  }
}

// ---------- agg conv1: 8 -> 16, WMMA; 9 row-chunks, const-immediate B loads ----------
__global__ void k_agg_conv1(const _Float16* __restrict__ cvp, const _Float16* __restrict__ ap,
                            _Float16* __restrict__ out, float* __restrict__ stats){
  __shared__ float ssum[8], sssq[8];
  if (threadIdx.x < 8){ ssum[threadIdx.x] = 0.f; sssq[threadIdx.x] = 0.f; }
  __syncthreads();
  int b = blockIdx.z / Dd, d = blockIdx.z % Dd, h = blockIdx.y;
  int lane = threadIdx.x & 31, wave = threadIdx.x >> 5;
  int w0 = blockIdx.x * 64 + wave * 16;
  int n = lane & 15;
  int laneHi = (lane & 16) ? 1 : 0;
  int sp8   = (((b*Dp + d)*Hp + h)*Wp + (w0 + n)) * 8;
  int sp8lo = sp8 + laneHi*8;                       // e[0..7]: kw = laneHi
  Frag8 accs[2];
  #pragma unroll
  for (int r = 0; r < 8; ++r){ accs[0].e[r] = 0.f; accs[1].e[r] = 0.f; }
  #pragma unroll
  for (int c = 0; c < 9; ++c){
    Frag16 a; a.v = *(const v16h*)(ap + (c*32 + lane)*16);
    const int rowOff = ((c/3)*Hp + (c%3))*Wp * 8;   // (kd*Hp + kh)*Wp in elems*8ch
    Frag16 bf;
    bf.h[0] = *(const v8h*)(cvp + sp8lo + rowOff);        // kw = laneHi
    bf.h[1] = *(const v8h*)(cvp + sp8   + rowOff + 16);   // kw = 2 (kw=3 slot has zero weight)
    accs[c & 1].v = __builtin_amdgcn_wmma_f32_16x16x32_f16(false, a.v, false, bf.v, (short)0, accs[c & 1].v, false, false);
  }
  Frag8 acc; acc.v = accs[0].v + accs[1].v;
  int cohalf = laneHi ? 8 : 0;
  int sp16 = (((b*Dd + d)*Hh + h)*Wn + (w0 + n)) * 16;
  v8h st;
  #pragma unroll
  for (int r = 0; r < 8; ++r) st[r] = (_Float16)acc.e[r];
  *(v8h*)(out + sp16 + cohalf) = st;
  #pragma unroll
  for (int t2 = 0; t2 < 4; ++t2){
    float x0 = acc.e[2*t2], x1 = acc.e[2*t2+1];
    atomicAdd(&ssum[t2 + laneHi*4], x0 + x1);
    atomicAdd(&sssq[t2 + laneHi*4], x0*x0 + x1*x1);
  }
  __syncthreads();
  if (threadIdx.x < 8){
    int bg = b*8 + threadIdx.x;
    atomicAdd(&stats[0*32 + bg*2 + 0], ssum[threadIdx.x]);
    atomicAdd(&stats[0*32 + bg*2 + 1], sssq[threadIdx.x]);
  }
}

// ---------- normalize+SiLU raw NHWC -> padded NHWC (halo written 0) ----------
__global__ void k_norm3d(const _Float16* __restrict__ src, _Float16* __restrict__ dst,
                         const float* __restrict__ stats, int stage,
                         const float* __restrict__ gamma, const float* __restrict__ beta){
  int t = blockIdx.x * 256 + threadIdx.x;
  if (t >= Wp*16) return;
  int c  = t & 15;
  int wx = t >> 4;
  int sp = blockIdx.y;
  int hy = sp % Hp; sp /= Hp;
  int dz = sp % Dp; int b = sp / Dp;
  _Float16 o = (_Float16)0.f;
  if (wx >= 1 && wx <= Wn && hy >= 1 && hy <= Hh && dz >= 1 && dz <= Dd){
    int bg = b*8 + (c >> 1);
    float mean = stats[128 + stage*32 + bg*2 + 0];
    float rstd = stats[128 + stage*32 + bg*2 + 1];
    float x  = (float)src[(((b*Dd + dz-1)*Hh + hy-1)*Wn + wx-1)*16 + c];
    float xn = (x - mean) * rstd * gamma[c] + beta[c];
    o = (_Float16)silu_f(xn);
  }
  dst[(blockIdx.y*Wp + wx)*16 + c] = o;
}

// ---------- agg conv2: 16 -> 16, WMMA; dual accumulators, const-immediate B loads ----------
__global__ void k_agg_conv2(const _Float16* __restrict__ hn, const _Float16* __restrict__ ap,
                            _Float16* __restrict__ out, float* __restrict__ stats){
  __shared__ float ssum[8], sssq[8];
  if (threadIdx.x < 8){ ssum[threadIdx.x] = 0.f; sssq[threadIdx.x] = 0.f; }
  __syncthreads();
  int b = blockIdx.z / Dd, d = blockIdx.z % Dd, h = blockIdx.y;
  int lane = threadIdx.x & 31, wave = threadIdx.x >> 5;
  int w0 = blockIdx.x * 64 + wave * 16;
  int n = lane & 15;
  int laneHi = (lane & 16) ? 1 : 0;
  int cib = laneHi * 8;
  int sp16 = ((((b*Dp + d)*Hp + h)*Wp + (w0 + n)) * 16) + cib;
  Frag8 accs[2];
  #pragma unroll
  for (int r = 0; r < 8; ++r){ accs[0].e[r] = 0.f; accs[1].e[r] = 0.f; }
  #pragma unroll
  for (int c = 0; c < 14; ++c){
    Frag16 a; a.v = *(const v16h*)(ap + (c*32 + lane)*16);
    const int offA = toff3(c*2) * 16;
    const int offB = toff3((c*2 + 1 > 26) ? 26 : c*2 + 1) * 16;
    Frag16 bf;
    bf.h[0] = *(const v8h*)(hn + sp16 + offA);
    bf.h[1] = *(const v8h*)(hn + sp16 + offB);
    accs[c & 1].v = __builtin_amdgcn_wmma_f32_16x16x32_f16(false, a.v, false, bf.v, (short)0, accs[c & 1].v, false, false);
  }
  Frag8 acc; acc.v = accs[0].v + accs[1].v;
  int cohalf = laneHi ? 8 : 0;
  int so = (((b*Dd + d)*Hh + h)*Wn + (w0 + n)) * 16;
  v8h st;
  #pragma unroll
  for (int r = 0; r < 8; ++r) st[r] = (_Float16)acc.e[r];
  *(v8h*)(out + so + cohalf) = st;
  #pragma unroll
  for (int t2 = 0; t2 < 4; ++t2){
    float x0 = acc.e[2*t2], x1 = acc.e[2*t2+1];
    atomicAdd(&ssum[t2 + laneHi*4], x0 + x1);
    atomicAdd(&sssq[t2 + laneHi*4], x0*x0 + x1*x1);
  }
  __syncthreads();
  if (threadIdx.x < 8){
    int bg = b*8 + threadIdx.x;
    atomicAdd(&stats[1*32 + bg*2 + 0], ssum[threadIdx.x]);
    atomicAdd(&stats[1*32 + bg*2 + 1], sssq[threadIdx.x]);
  }
}

// ---------- agg conv3: 16 -> 1 ; one v16h load per tap ----------
__global__ void k_agg_conv3(const _Float16* __restrict__ hn, const float* __restrict__ w3,
                            const float* __restrict__ bias3, float* __restrict__ logits){
  int idx = blockIdx.x * 256 + threadIdx.x;
  if (idx >= Bn*Dd*HWn) return;
  int w = idx % Wn, h = (idx / Wn) % Hh, d = (idx / HWn) % Dd, b = idx / (Dd*HWn);
  int base = (((b*Dp + d)*Hp + h)*Wp + w) * 16;
  float acc = bias3[0];
  #pragma unroll
  for (int t = 0; t < 27; ++t){
    Frag16 x; x.v = *(const v16h*)(hn + base + toff3(t)*16);
    #pragma unroll
    for (int ci = 0; ci < 16; ++ci)
      acc += (float)x.e[ci] * w3[ci*27 + t];
  }
  logits[idx] = acc;
}

// ---------- softmax / soft-argmin / conf ; writes sx,sy = 0 ----------
__global__ void k_softmax(const float* __restrict__ logits, float* __restrict__ dsa,
                          float* __restrict__ dout){
  int idx = blockIdx.x * 256 + threadIdx.x;
  if (idx >= Bn*HWn) return;
  int b = idx / HWn, hw = idx % HWn;
  const float* lp = logits + (b*Dd)*HWn + hw;
  float m = -1e30f;
  for (int d = 0; d < Dd; ++d) m = fmaxf(m, lp[d*HWn]);
  float s = 0.f, wsum = 0.f, pm = 0.f;
  for (int d = 0; d < Dd; ++d){
    float e = __expf(lp[d*HWn] - m);
    s += e; wsum += e * (float)d; pm = fmaxf(pm, e);
  }
  float inv = 1.f / s;
  dsa[idx] = wsum * inv;
  dout[1945600 + idx] = pm * inv;   // conf
  dout[102400  + idx] = 0.f;        // sx
  dout[204800  + idx] = 0.f;        // sy
}

// ---------- feature head conv: 64 -> 16, WMMA; dual accumulators ----------
__global__ void k_feat_conv(const _Float16* __restrict__ fLp, const _Float16* __restrict__ ap,
                            float* __restrict__ outraw, float* __restrict__ stats){
  __shared__ float ssum[8], sssq[8];
  if (threadIdx.x < 8){ ssum[threadIdx.x] = 0.f; sssq[threadIdx.x] = 0.f; }
  __syncthreads();
  int b = blockIdx.z, h = blockIdx.y;
  int lane = threadIdx.x & 31, wave = threadIdx.x >> 5;
  int w0 = blockIdx.x * 64 + wave * 16;
  int n = lane & 15;
  int laneHi = (lane & 16) ? 1 : 0;
  int sp64 = (((b*Hp + h)*Wp + (w0 + n)) * 64) + laneHi*8;
  Frag8 accs[2];
  #pragma unroll
  for (int r = 0; r < 8; ++r){ accs[0].e[r] = 0.f; accs[1].e[r] = 0.f; }
  #pragma unroll
  for (int c = 0; c < 18; ++c){
    Frag16 a; a.v = *(const v16h*)(ap + (c*32 + lane)*16);
    const int cof = (c & 1)*32 + toff2(c >> 1)*64;
    Frag16 bf;
    bf.h[0] = *(const v8h*)(fLp + sp64 + cof);
    bf.h[1] = *(const v8h*)(fLp + sp64 + cof + 16);
    accs[c & 1].v = __builtin_amdgcn_wmma_f32_16x16x32_f16(false, a.v, false, bf.v, (short)0, accs[c & 1].v, false, false);
  }
  Frag8 acc; acc.v = accs[0].v + accs[1].v;
  int cohalf = laneHi ? 8 : 0;
  int so = ((b*Hh + h)*Wn + (w0 + n)) * 16;
  *(v8f*)(outraw + so + cohalf) = acc.v;
  #pragma unroll
  for (int t2 = 0; t2 < 4; ++t2){
    float x0 = acc.e[2*t2], x1 = acc.e[2*t2+1];
    atomicAdd(&ssum[t2 + laneHi*4], x0 + x1);
    atomicAdd(&sssq[t2 + laneHi*4], x0*x0 + x1*x1);
  }
  __syncthreads();
  if (threadIdx.x < 8){
    int bg = b*8 + threadIdx.x;
    atomicAdd(&stats[2*32 + bg*2 + 0], ssum[threadIdx.x]);
    atomicAdd(&stats[2*32 + bg*2 + 1], sssq[threadIdx.x]);
  }
}

// ---------- normalize+SiLU feat: NHWC f32 raw -> NCHW f32 output ----------
__global__ void k_norm2d(const float* __restrict__ src, float* __restrict__ dst,
                         const float* __restrict__ stats, int stage,
                         const float* __restrict__ gamma, const float* __restrict__ beta){
  int idx = blockIdx.x * 256 + threadIdx.x;
  if (idx >= Bn*16*HWn) return;
  int c = (idx / HWn) % 16;
  int b = idx / (16*HWn);
  int hw = idx % HWn;
  int bg = b*8 + (c >> 1);
  float mean = stats[128 + stage*32 + bg*2 + 0];
  float rstd = stats[128 + stage*32 + bg*2 + 1];
  float xn = (src[(b*HWn + hw)*16 + c] - mean) * rstd * gamma[c] + beta[c];
  dst[idx] = silu_f(xn);
}

// ---------- apc: conv1d over flattened H*W, 24 -> 24, k=3 ----------
__global__ void k_apc(const float* __restrict__ logits, const float* __restrict__ wv,
                      float* __restrict__ outp){
  int idx = blockIdx.x * 256 + threadIdx.x;
  if (idx >= Bn*Dd*HWn) return;
  int p = idx % HWn, co = (idx / HWn) % Dd, b = idx / (Dd*HWn);
  float acc = 0.f;
  for (int ci = 0; ci < Dd; ++ci){
    const float* L = logits + (b*Dd + ci)*HWn;
    const float* wp = wv + (co*Dd + ci)*3;
    #pragma unroll
    for (int t = 0; t < 3; ++t){
      int pp = p + t - 1;
      if ((unsigned)pp < (unsigned)HWn) acc += L[pp] * wp[t];
    }
  }
  outp[idx] = acc;
}

// ---------- regression conv1: (24 apc + 16 feat) -> 32, 3x3 (VALU) ----------
__global__ void k_reg1(const float* __restrict__ apc, const float* __restrict__ featn,
                       const float* __restrict__ wv, float* __restrict__ outp,
                       float* __restrict__ stats){
  __shared__ float red[256], red2[256];
  int tid = threadIdx.x;
  int idx = blockIdx.x * 256 + tid;       // grid sized exactly: Bn*32*HWn / 256
  int hw = idx % HWn, co = (idx / HWn) % 32, b = idx / (32*HWn);
  int h = hw / Wn, w = hw % Wn;
  float acc = 0.f;
  for (int ic = 0; ic < 40; ++ic){
    const float* src = (ic < 24) ? (apc + (b*Dd + ic)*HWn) : (featn + (b*16 + (ic - 24))*HWn);
    const float* wp = wv + (co*40 + ic)*9;
    #pragma unroll
    for (int kh = 0; kh < 3; ++kh){
      int hh = h + kh - 1;
      if ((unsigned)hh >= (unsigned)Hh) continue;
      #pragma unroll
      for (int kw = 0; kw < 3; ++kw){
        int ww = w + kw - 1;
        if ((unsigned)ww < (unsigned)Wn) acc += src[hh*Wn + ww] * wp[kh*3 + kw];
      }
    }
  }
  outp[idx] = acc;
  red[tid] = acc; red2[tid] = acc*acc;
  __syncthreads();
  for (int s = 128; s > 0; s >>= 1){
    if (tid < s){ red[tid] += red[tid + s]; red2[tid] += red2[tid + s]; }
    __syncthreads();
  }
  if (tid == 0){
    int bg = b*8 + (co >> 2);
    atomicAdd(&stats[3*32 + bg*2 + 0], red[0]);
    atomicAdd(&stats[3*32 + bg*2 + 1], red2[0]);
  }
}

// ---------- regression conv2: 32 -> 1 with on-the-fly GN+SiLU, write d_init ----------
__global__ void k_reg2(const float* __restrict__ h3, const float* __restrict__ stats,
                       const float* __restrict__ g, const float* __restrict__ be,
                       const float* __restrict__ wv, const float* __restrict__ bias,
                       const float* __restrict__ dsa, float* __restrict__ dout){
  int idx = blockIdx.x * 256 + threadIdx.x;
  if (idx >= Bn*HWn) return;
  int b = idx / HWn, hw = idx % HWn;
  int h = hw / Wn, w = hw % Wn;
  float acc = bias[0];
  for (int ci = 0; ci < 32; ++ci){
    int bg = b*8 + (ci >> 2);
    float mean = stats[128 + 3*32 + bg*2 + 0];
    float rstd = stats[128 + 3*32 + bg*2 + 1];
    float ga = g[ci], bb = be[ci];
    const float* src = h3 + (b*32 + ci)*HWn;
    const float* wp = wv + ci*9;
    #pragma unroll
    for (int kh = 0; kh < 3; ++kh){
      int hh = h + kh - 1;
      if ((unsigned)hh >= (unsigned)Hh) continue;
      #pragma unroll
      for (int kw = 0; kw < 3; ++kw){
        int ww = w + kw - 1;
        if ((unsigned)ww >= (unsigned)Wn) continue;
        float xn = (src[hh*Wn + ww] - mean) * rstd * ga + bb;
        acc += silu_f(xn) * wp[kh*3 + kw];
      }
    }
  }
  dout[idx] = dsa[idx] + acc;  // d_init
}

// ---------- host ----------
extern "C" void kernel_launch(void* const* d_in, const int* in_sizes, int n_in,
                              void* d_out, int out_size, void* d_ws, size_t ws_size,
                              hipStream_t stream) {
  (void)in_sizes; (void)n_in; (void)out_size; (void)ws_size;
  const float* fL    = (const float*)d_in[0];
  const float* fR    = (const float*)d_in[1];
  const float* aw1   = (const float*)d_in[2];
  const float* ag1   = (const float*)d_in[3];
  const float* ab1   = (const float*)d_in[4];
  const float* aw2   = (const float*)d_in[5];
  const float* ag2   = (const float*)d_in[6];
  const float* ab2   = (const float*)d_in[7];
  const float* aw3   = (const float*)d_in[8];
  const float* ab3   = (const float*)d_in[9];
  const float* fw    = (const float*)d_in[10];
  const float* fg    = (const float*)d_in[11];
  const float* fb    = (const float*)d_in[12];
  const float* apcw  = (const float*)d_in[13];
  const float* rw1   = (const float*)d_in[14];
  const float* rg1   = (const float*)d_in[15];
  const float* rb1   = (const float*)d_in[16];
  const float* rw2   = (const float*)d_in[17];
  const float* rb2   = (const float*)d_in[18];
  float* dout = (float*)d_out;

  char* ws = (char*)d_ws;
  size_t off = 0;
  auto take = [&](size_t bytes) -> void* {
    void* p = ws + off;
    off = (off + bytes + 255) & ~(size_t)255;
    return p;
  };
  float*     stats   = (float*)    take(256ull * 4);
  _Float16*  cvp     = (_Float16*) take((size_t)Bn*SP3*8 * 2);    // padded NHWC cv (8 ch)
  _Float16*  h1raw   = (_Float16*) take(39321600ull * 2);         // NHWC raw (16 ch)
  _Float16*  h2raw   = (_Float16*) take(39321600ull * 2);
  _Float16*  hnp     = (_Float16*) take((size_t)Bn*SP3*16 * 2);   // padded NHWC normalized
  _Float16*  fLp     = (_Float16*) take((size_t)Bn*SP2*64 * 2);   // padded NHWC f16 fL
  _Float16*  ap1     = (_Float16*) take(4608ull * 2);
  _Float16*  ap2     = (_Float16*) take(7168ull * 2);
  _Float16*  apF     = (_Float16*) take(9216ull * 2);
  float*     logits  = (float*)    take(2457600ull * 4);
  float*     dsa     = (float*)    take(102400ull * 4);
  float*     featraw = (float*)    take(1638400ull * 4);          // NHWC f32 (16 ch)
  float*     apcout  = (float*)    take(2457600ull * 4);
  float*     h3raw   = (float*)    take(3276800ull * 4);

  float* featOut = dout + 307200;

  const int cvWords  = Bn*SP3*8/2;
  const int fLpWords = Bn*SP2*64/2;

  k_zero<<<1, 256, 0, stream>>>(stats, 256);
  k_pack_w1<<<1, 256, 0, stream>>>(aw1, ap1);
  k_pack_w2<<<1, 256, 0, stream>>>(aw2, ap2);
  k_pack_wf<<<1, 256, 0, stream>>>(fw, apF);
  k_fill32<<<(cvWords + 255)/256, 256, 0, stream>>>((unsigned int*)cvp, cvWords);
  k_fill32<<<(fLpWords + 255)/256, 256, 0, stream>>>((unsigned int*)fLp, fLpWords);

  k_padLt<<<dim3(5, Hh, Bn), 256, 0, stream>>>(fL, fLp);
  k_cost<<<dim3(5, Hh, Bn), 256, 0, stream>>>(fL, fR, cvp);

  dim3 ngrid((Wp*16 + 255)/256, Bn*Dp*Hp, 1);   // x: Wp*16 elems, y: (b,dz,hy)

  k_agg_conv1<<<dim3(5, Hh, Bn*Dd), 128, 0, stream>>>(cvp, ap1, h1raw, stats);
  k_finalize<<<1, 64, 0, stream>>>(stats);
  k_norm3d<<<ngrid, 256, 0, stream>>>(h1raw, hnp, stats, 0, ag1, ab1);

  k_agg_conv2<<<dim3(5, Hh, Bn*Dd), 128, 0, stream>>>(hnp, ap2, h2raw, stats);
  k_finalize<<<1, 64, 0, stream>>>(stats);
  k_norm3d<<<ngrid, 256, 0, stream>>>(h2raw, hnp, stats, 1, ag2, ab2);

  k_agg_conv3<<<9600, 256, 0, stream>>>(hnp, aw3, ab3, logits);
  k_softmax<<<400, 256, 0, stream>>>(logits, dsa, dout);

  k_feat_conv<<<dim3(5, Hh, Bn), 128, 0, stream>>>(fLp, apF, featraw, stats);
  k_finalize<<<1, 64, 0, stream>>>(stats);
  k_norm2d<<<6400, 256, 0, stream>>>(featraw, featOut, stats, 2, fg, fb);

  k_apc<<<9600, 256, 0, stream>>>(logits, apcw, apcout);

  k_reg1<<<12800, 256, 0, stream>>>(apcout, featOut, rw1, h3raw, stats);
  k_finalize<<<1, 64, 0, stream>>>(stats);
  k_reg2<<<400, 256, 0, stream>>>(h3raw, stats, rg1, rb1, rw2, rb2, dsa, dout);
}
